// ExactSlidingWindowAttention_23089744183532
// MI455X (gfx1250) — matrix operand order repaired and checked by reading, not compiled, not run on hardware
//
#include <hip/hip_runtime.h>
#include <math.h>

// Problem constants (from reference): B=16, S=4096, D=64, window=512, causal.
#define B_  16
#define S_  4096
#define D_  64
#define W_  512

typedef __attribute__((ext_vector_type(16))) __bf16 v16bf;
typedef __attribute__((ext_vector_type(8)))  __bf16 v8bf;
typedef __attribute__((ext_vector_type(2)))  __bf16 v2bf;
typedef __attribute__((ext_vector_type(8)))  float  v8f;

// ---------------------------------------------------------------------------
// Prep pass 1: Q -> bf16 (pre-scaled by 1/sqrt(D)), K -> bf16. Elementwise.
// ---------------------------------------------------------------------------
__global__ __launch_bounds__(256)
void swa_prep_qk(const float* __restrict__ Q, const float* __restrict__ K,
                 __bf16* __restrict__ Qs, __bf16* __restrict__ Ks) {
    const size_t i = (size_t)(blockIdx.x * blockDim.x + threadIdx.x) * 2;
    const float2 q = *(const float2*)(Q + i);
    const float2 k = *(const float2*)(K + i);
    v2bf qb, kb;
    qb[0] = (__bf16)(q.x * 0.125f); qb[1] = (__bf16)(q.y * 0.125f);
    kb[0] = (__bf16)k.x;            kb[1] = (__bf16)k.y;
    *(v2bf*)(Qs + i) = qb;
    *(v2bf*)(Ks + i) = kb;
}

// ---------------------------------------------------------------------------
// Prep pass 2: V[b][s][d] -> Vt[b][d][s] in bf16 (tiled LDS transpose).
// ---------------------------------------------------------------------------
__global__ __launch_bounds__(256)
void swa_prep_vt(const float* __restrict__ V, __bf16* __restrict__ Vt) {
    __shared__ float tile[32][33];
    const int dTiles = D_ / 32;          // 2
    const int sTiles = S_ / 32;          // 128
    const int b  = blockIdx.x / (sTiles * dTiles);
    const int r  = blockIdx.x % (sTiles * dTiles);
    const int st = r / dTiles;
    const int dt = r % dTiles;
    const int tx = threadIdx.x & 31;
    const int ty = threadIdx.x >> 5;     // 0..7

    #pragma unroll
    for (int k = 0; k < 4; ++k) {
        const int s = st * 32 + ty + 8 * k;
        tile[ty + 8 * k][tx] = V[((size_t)b * S_ + s) * D_ + dt * 32 + tx];
    }
    __syncthreads();
    #pragma unroll
    for (int k = 0; k < 4; ++k) {
        const int d = dt * 32 + ty + 8 * k;
        Vt[((size_t)b * D_ + d) * S_ + st * 32 + tx] = (__bf16)tile[tx][ty + 8 * k];
    }
}

// ---------------------------------------------------------------------------
// Main flash-attention kernel. 64 keys / iteration (16 WMMAs), wave-uniform
// fast path for fully-in-window key blocks.
// ---------------------------------------------------------------------------
__global__ __launch_bounds__(128)
void ExactSlidingWindowAttention_23089744183532_kernel(const __bf16* __restrict__ Qs,
                                                       const __bf16* __restrict__ Ks,
                                                       const __bf16* __restrict__ Vt,
                                                       float* __restrict__ O) {
    __shared__ __bf16 lds_p[4][16][64];   // per-wave P transpose buffer (8 KB)

    const int wave = threadIdx.x >> 5;
    const int lane = threadIdx.x & 31;
    const int half = lane >> 4;
    const int ln   = lane & 15;

    const int qblocks = S_ / 64;
    const int b   = blockIdx.x / qblocks;
    const int q0  = (blockIdx.x % qblocks) * 64;
    const int qt0 = q0 + wave * 16;

    const __bf16* Qb  = Qs + (size_t)b * S_ * D_;
    const __bf16* Kb  = Ks + (size_t)b * S_ * D_;
    const __bf16* Vtb = Vt + (size_t)b * D_ * S_;
    float*        Ob  = O  + (size_t)b * S_ * D_;

    // ---- Q tile in A layout ----
    v16bf qa0, qa1;
    {
        const v8bf* qc = (const v8bf*)(Qb + (size_t)(qt0 + ln) * D_);
        v8bf c0 = qc[2 * half], c1 = qc[2 * half + 1], c2 = qc[4 + 2 * half], c3 = qc[5 + 2 * half];
        qa0 = __builtin_shufflevector(c0, c1, 0,1,2,3,4,5,6,7,8,9,10,11,12,13,14,15);
        qa1 = __builtin_shufflevector(c2, c3, 0,1,2,3,4,5,6,7,8,9,10,11,12,13,14,15);
    }

    v8f acc0 = {}, acc1 = {}, acc2 = {}, acc3 = {};
    float mrow[8], lpart[8];
    #pragma unroll
    for (int r = 0; r < 8; ++r) { mrow[r] = -3.0e38f; lpart[r] = 0.0f; }

    int klo = q0 - (W_ - 1); if (klo < 0) klo = 0;
    klo &= ~63;
    const int khi = q0 + 63;   // inclusive

    for (int kb = klo; kb <= khi; kb += 64) {
        // ---- K in B layout: 4 key tiles x 2 D halves ----
        const __bf16* kr0 = Kb + (size_t)(kb + ln) * D_ + half * 16;
        const __bf16* kr1 = kr0 + (size_t)16 * D_;
        const __bf16* kr2 = kr0 + (size_t)32 * D_;
        const __bf16* kr3 = kr0 + (size_t)48 * D_;
        v16bf bk0a = *(const v16bf*)kr0, bk0b = *(const v16bf*)(kr0 + 32);
        v16bf bk1a = *(const v16bf*)kr1, bk1b = *(const v16bf*)(kr1 + 32);
        v16bf bk2a = *(const v16bf*)kr2, bk2b = *(const v16bf*)(kr2 + 32);
        v16bf bk3a = *(const v16bf*)kr3, bk3b = *(const v16bf*)(kr3 + 32);

        if (kb + 64 <= khi) __builtin_prefetch(kr0 + (size_t)64 * D_, 0, 1);

        v8f c0 = {}, c1 = {}, c2 = {}, c3 = {};
        c0 = __builtin_amdgcn_wmma_f32_16x16x32_bf16(false, qa0, false, bk0a, (short)0, c0, false, false);
        c0 = __builtin_amdgcn_wmma_f32_16x16x32_bf16(false, qa1, false, bk0b, (short)0, c0, false, false);
        c1 = __builtin_amdgcn_wmma_f32_16x16x32_bf16(false, qa0, false, bk1a, (short)0, c1, false, false);
        c1 = __builtin_amdgcn_wmma_f32_16x16x32_bf16(false, qa1, false, bk1b, (short)0, c1, false, false);
        c2 = __builtin_amdgcn_wmma_f32_16x16x32_bf16(false, qa0, false, bk2a, (short)0, c2, false, false);
        c2 = __builtin_amdgcn_wmma_f32_16x16x32_bf16(false, qa1, false, bk2b, (short)0, c2, false, false);
        c3 = __builtin_amdgcn_wmma_f32_16x16x32_bf16(false, qa0, false, bk3a, (short)0, c3, false, false);
        c3 = __builtin_amdgcn_wmma_f32_16x16x32_bf16(false, qa1, false, bk3b, (short)0, c3, false, false);

        // Fully-valid block for this wave? (all j<=i and j>=i-511)
        const bool unmasked = (kb >= qt0 - (W_ - 16)) && (kb <= qt0 - 63);

        if (unmasked) {
            #pragma unroll
            for (int r = 0; r < 8; ++r) {
                float mx = fmaxf(fmaxf(c0[r], c1[r]), fmaxf(c2[r], c3[r]));
                #pragma unroll
                for (int off = 1; off < 16; off <<= 1)
                    mx = fmaxf(mx, __shfl_xor(mx, off, 32));
                const float mnew  = fmaxf(mrow[r], mx);
                const float alpha = __expf(mrow[r] - mnew);
                const float p0 = __expf(c0[r] - mnew);
                const float p1 = __expf(c1[r] - mnew);
                const float p2 = __expf(c2[r] - mnew);
                const float p3 = __expf(c3[r] - mnew);
                lpart[r] = lpart[r] * alpha + ((p0 + p1) + (p2 + p3));
                mrow[r]  = mnew;
                acc0[r] *= alpha; acc1[r] *= alpha; acc2[r] *= alpha; acc3[r] *= alpha;
                __bf16* prow = &lds_p[wave][r + 8 * half][0];
                prow[ln]      = (__bf16)p0;
                prow[16 + ln] = (__bf16)p1;
                prow[32 + ln] = (__bf16)p2;
                prow[48 + ln] = (__bf16)p3;
            }
        } else {
            #pragma unroll
            for (int r = 0; r < 8; ++r) {
                const int i  = qt0 + r + 8 * half;
                const int j0 = kb + ln;
                const bool v0 = (j0      <= i) && (j0      >= i - (W_ - 1));
                const bool v1 = (j0 + 16 <= i) && (j0 + 16 >= i - (W_ - 1));
                const bool v2 = (j0 + 32 <= i) && (j0 + 32 >= i - (W_ - 1));
                const bool v3 = (j0 + 48 <= i) && (j0 + 48 >= i - (W_ - 1));
                const float s0 = v0 ? c0[r] : -3.0e38f;
                const float s1 = v1 ? c1[r] : -3.0e38f;
                const float s2 = v2 ? c2[r] : -3.0e38f;
                const float s3 = v3 ? c3[r] : -3.0e38f;
                float mx = fmaxf(fmaxf(s0, s1), fmaxf(s2, s3));
                #pragma unroll
                for (int off = 1; off < 16; off <<= 1)
                    mx = fmaxf(mx, __shfl_xor(mx, off, 32));
                const float mnew  = fmaxf(mrow[r], mx);
                const float alpha = __expf(mrow[r] - mnew);
                const float p0 = v0 ? __expf(s0 - mnew) : 0.0f;
                const float p1 = v1 ? __expf(s1 - mnew) : 0.0f;
                const float p2 = v2 ? __expf(s2 - mnew) : 0.0f;
                const float p3 = v3 ? __expf(s3 - mnew) : 0.0f;
                lpart[r] = lpart[r] * alpha + ((p0 + p1) + (p2 + p3));
                mrow[r]  = mnew;
                acc0[r] *= alpha; acc1[r] *= alpha; acc2[r] *= alpha; acc3[r] *= alpha;
                __bf16* prow = &lds_p[wave][r + 8 * half][0];
                prow[ln]      = (__bf16)p0;
                prow[16 + ln] = (__bf16)p1;
                prow[32 + ln] = (__bf16)p2;
                prow[48 + ln] = (__bf16)p3;
            }
        }
        __syncthreads();

        // ---- P back as A operands (two 16x32 chunks) ----
        v16bf pa0, pa1;
        {
            const v8bf* prow = (const v8bf*)&lds_p[wave][ln][0];
            v8bf a = prow[2 * half], bch = prow[2 * half + 1];
            v8bf c = prow[4 + 2 * half], d = prow[5 + 2 * half];
            pa0 = __builtin_shufflevector(a, bch, 0,1,2,3,4,5,6,7,8,9,10,11,12,13,14,15);
            pa1 = __builtin_shufflevector(c, d,   0,1,2,3,4,5,6,7,8,9,10,11,12,13,14,15);
        }
        __syncthreads();

        // ---- V^T in B layout: 4 d-tiles x 2 key chunks ----
        const __bf16* vtc = Vtb + (size_t)ln * S_ + kb + half * 16;
        v16bf vb00 = *(const v16bf*)(vtc);
        v16bf vb01 = *(const v16bf*)(vtc + 32);
        v16bf vb10 = *(const v16bf*)(vtc + (size_t)16 * S_);
        v16bf vb11 = *(const v16bf*)(vtc + (size_t)16 * S_ + 32);
        v16bf vb20 = *(const v16bf*)(vtc + (size_t)32 * S_);
        v16bf vb21 = *(const v16bf*)(vtc + (size_t)32 * S_ + 32);
        v16bf vb30 = *(const v16bf*)(vtc + (size_t)48 * S_);
        v16bf vb31 = *(const v16bf*)(vtc + (size_t)48 * S_ + 32);

        if (kb + 64 <= khi) __builtin_prefetch(vtc + 64, 0, 1);

        acc0 = __builtin_amdgcn_wmma_f32_16x16x32_bf16(false, pa0, false, vb00, (short)0, acc0, false, false);
        acc0 = __builtin_amdgcn_wmma_f32_16x16x32_bf16(false, pa1, false, vb01, (short)0, acc0, false, false);
        acc1 = __builtin_amdgcn_wmma_f32_16x16x32_bf16(false, pa0, false, vb10, (short)0, acc1, false, false);
        acc1 = __builtin_amdgcn_wmma_f32_16x16x32_bf16(false, pa1, false, vb11, (short)0, acc1, false, false);
        acc2 = __builtin_amdgcn_wmma_f32_16x16x32_bf16(false, pa0, false, vb20, (short)0, acc2, false, false);
        acc2 = __builtin_amdgcn_wmma_f32_16x16x32_bf16(false, pa1, false, vb21, (short)0, acc2, false, false);
        acc3 = __builtin_amdgcn_wmma_f32_16x16x32_bf16(false, pa0, false, vb30, (short)0, acc3, false, false);
        acc3 = __builtin_amdgcn_wmma_f32_16x16x32_bf16(false, pa1, false, vb31, (short)0, acc3, false, false);
    }

    // ---- Final row-sum reduction, normalize, store ----
    #pragma unroll
    for (int r = 0; r < 8; ++r) {
        float ls = lpart[r];
        #pragma unroll
        for (int off = 1; off < 16; off <<= 1)
            ls += __shfl_xor(ls, off, 32);
        const float inv = 1.0f / ls;
        float* orow = Ob + (size_t)(qt0 + r + 8 * half) * D_ + ln;
        orow[0]  = acc0[r] * inv;
        orow[16] = acc1[r] * inv;
        orow[32] = acc2[r] * inv;
        orow[48] = acc3[r] * inv;
    }
}

// ---------------------------------------------------------------------------
// Fallback (reads f32 directly) if workspace too small.
// ---------------------------------------------------------------------------
__global__ __launch_bounds__(128)
void swa_fallback_f32(const float* __restrict__ Q, const float* __restrict__ K,
                      const float* __restrict__ V, float* __restrict__ O) {
    __shared__ __bf16 lds_p[4][16][32];
    const int wave = threadIdx.x >> 5, lane = threadIdx.x & 31;
    const int half = lane >> 4, ln = lane & 15;
    const int qblocks = S_ / 64;
    const int b = blockIdx.x / qblocks, q0 = (blockIdx.x % qblocks) * 64;
    const int qt0 = q0 + wave * 16;
    const float scale = 0.125f;
    const float* Qb = Q + (size_t)b * S_ * D_;
    const float* Kb = K + (size_t)b * S_ * D_;
    const float* Vb = V + (size_t)b * S_ * D_;
    float* Ob = O + (size_t)b * S_ * D_;
    v16bf qa0, qa1;
    {
        const float* qrow = Qb + (size_t)(qt0 + ln) * D_;
        #pragma unroll
        for (int e = 0; e < 16; ++e) {
            const int d = e + half * 16;
            qa0[e] = (__bf16)(qrow[d] * scale);
            qa1[e] = (__bf16)(qrow[d + 32] * scale);
        }
    }
    v8f acc0 = {}, acc1 = {}, acc2 = {}, acc3 = {};
    float mrow[8], lrow[8];
    #pragma unroll
    for (int r = 0; r < 8; ++r) { mrow[r] = -3.0e38f; lrow[r] = 0.0f; }
    int klo = q0 - (W_ - 1); if (klo < 0) klo = 0;
    klo &= ~31;
    const int khi = q0 + 63;
    for (int kb = klo; kb <= khi; kb += 32) {
        v16bf bk00, bk01, bk10, bk11;
        {
            const float* kr0 = Kb + (size_t)(kb + ln) * D_ + half * 16;
            const float* kr1 = Kb + (size_t)(kb + 16 + ln) * D_ + half * 16;
            #pragma unroll
            for (int e = 0; e < 16; ++e) {
                bk00[e] = (__bf16)kr0[e];  bk01[e] = (__bf16)kr0[e + 32];
                bk10[e] = (__bf16)kr1[e];  bk11[e] = (__bf16)kr1[e + 32];
            }
        }
        v8f c0 = {}, c1 = {};
        c0 = __builtin_amdgcn_wmma_f32_16x16x32_bf16(false, qa0, false, bk00, (short)0, c0, false, false);
        c0 = __builtin_amdgcn_wmma_f32_16x16x32_bf16(false, qa1, false, bk01, (short)0, c0, false, false);
        c1 = __builtin_amdgcn_wmma_f32_16x16x32_bf16(false, qa0, false, bk10, (short)0, c1, false, false);
        c1 = __builtin_amdgcn_wmma_f32_16x16x32_bf16(false, qa1, false, bk11, (short)0, c1, false, false);
        #pragma unroll
        for (int r = 0; r < 8; ++r) {
            const int i = qt0 + r + 8 * half;
            const int j0 = kb + ln, j1 = kb + 16 + ln;
            const bool v0 = (j0 <= i) && (j0 >= i - (W_ - 1));
            const bool v1 = (j1 <= i) && (j1 >= i - (W_ - 1));
            const float s0 = v0 ? c0[r] : -3.0e38f;
            const float s1 = v1 ? c1[r] : -3.0e38f;
            float mx = fmaxf(s0, s1);
            #pragma unroll
            for (int off = 1; off < 16; off <<= 1)
                mx = fmaxf(mx, __shfl_xor(mx, off, 32));
            const float mnew = fmaxf(mrow[r], mx);
            const float alpha = __expf(mrow[r] - mnew);
            const float p0 = v0 ? __expf(s0 - mnew) : 0.0f;
            const float p1 = v1 ? __expf(s1 - mnew) : 0.0f;
            float rs = p0 + p1;
            #pragma unroll
            for (int off = 1; off < 16; off <<= 1)
                rs += __shfl_xor(rs, off, 32);
            lrow[r] = lrow[r] * alpha + rs;
            mrow[r] = mnew;
            acc0[r] *= alpha; acc1[r] *= alpha; acc2[r] *= alpha; acc3[r] *= alpha;
            lds_p[wave][r + 8 * half][ln]      = (__bf16)p0;
            lds_p[wave][r + 8 * half][16 + ln] = (__bf16)p1;
        }
        __syncthreads();
        v16bf pa;
        #pragma unroll
        for (int e = 0; e < 16; ++e) {
            const int kc = e + half * 16;
            pa[e] = lds_p[wave][ln][kc];
        }
        __syncthreads();
        v16bf vb0, vb1, vb2, vb3;
        {
            const float* vc = Vb + (size_t)(kb + half * 16) * D_ + ln;
            #pragma unroll
            for (int e = 0; e < 16; ++e) {
                const float* vr = vc + (size_t)e * D_;
                vb0[e] = (__bf16)vr[0];  vb1[e] = (__bf16)vr[16];
                vb2[e] = (__bf16)vr[32]; vb3[e] = (__bf16)vr[48];
            }
        }
        acc0 = __builtin_amdgcn_wmma_f32_16x16x32_bf16(false, pa, false, vb0, (short)0, acc0, false, false);
        acc1 = __builtin_amdgcn_wmma_f32_16x16x32_bf16(false, pa, false, vb1, (short)0, acc1, false, false);
        acc2 = __builtin_amdgcn_wmma_f32_16x16x32_bf16(false, pa, false, vb2, (short)0, acc2, false, false);
        acc3 = __builtin_amdgcn_wmma_f32_16x16x32_bf16(false, pa, false, vb3, (short)0, acc3, false, false);
    }
    #pragma unroll
    for (int r = 0; r < 8; ++r) {
        const float inv = 1.0f / lrow[r];
        float* orow = Ob + (size_t)(qt0 + r + 8 * half) * D_ + ln;
        orow[0]  = acc0[r] * inv;  orow[16] = acc1[r] * inv;
        orow[32] = acc2[r] * inv;  orow[48] = acc3[r] * inv;
    }
}

extern "C" void kernel_launch(void* const* d_in, const int* in_sizes, int n_in,
                              void* d_out, int out_size, void* d_ws, size_t ws_size,
                              hipStream_t stream) {
    (void)in_sizes; (void)n_in; (void)out_size;
    const float* q = (const float*)d_in[0];
    const float* k = (const float*)d_in[1];
    const float* v = (const float*)d_in[2];
    float* o = (float*)d_out;

    const size_t N      = (size_t)B_ * S_ * D_;          // 4,194,304 elems
    const size_t need   = 3 * N * sizeof(__bf16);        // ~25.2 MB
    const int    blocks = B_ * (S_ / 64);                // 1024 workgroups

    if (ws_size >= need) {
        __bf16* Qs = (__bf16*)d_ws;
        __bf16* Ks = Qs + N;
        __bf16* Vt = Ks + N;
        swa_prep_qk<<<(int)(N / 2 / 256), 256, 0, stream>>>(q, k, Qs, Ks);
        swa_prep_vt<<<B_ * (S_ / 32) * (D_ / 32), 256, 0, stream>>>(v, Vt);
        ExactSlidingWindowAttention_23089744183532_kernel<<<blocks, 128, 0, stream>>>(Qs, Ks, Vt, o);
    } else {
        swa_fallback_f32<<<blocks, 128, 0, stream>>>(q, k, v, o);
    }
}
